// SelectiveLookups_41893111005223
// MI455X (gfx1250) — matrix operand order repaired and checked by reading, not compiled, not run on hardware
//
#include <hip/hip_runtime.h>
#include <math.h>

#define BB 2
#define SS 2048
#define DD 1024
#define HH 16
#define HDIM 64
#define KTOP 46   // ceil(sqrt(2048))

typedef __attribute__((ext_vector_type(16))) __bf16 v16bf;
typedef __attribute__((ext_vector_type(8)))  __bf16 v8bf;
typedef __attribute__((ext_vector_type(8)))  float  v8f;

// ---------- scalar bf16 helpers (round-to-nearest-even) ----------
static __device__ __forceinline__ __bf16 f2bf(float f) {
  union { float f; unsigned u; } cv; cv.f = f;
  unsigned u = cv.u;
  u = (u + 0x7FFFu + ((u >> 16) & 1u)) >> 16;
  union { unsigned short s; __bf16 b; } o; o.s = (unsigned short)u;
  return o.b;
}

static __device__ __forceinline__ v16bf make16(v8bf lo, v8bf hi) {
  v16bf r;
#pragma unroll
  for (int i = 0; i < 8; i++) { r[i] = lo[i]; r[i + 8] = hi[i]; }
  return r;
}

static __device__ __forceinline__ v8f wmma_bf16(v16bf a, v16bf b, v8f c) {
  // (neg_a, A, neg_b, B, c_mod, C, reuse_a, reuse_b)
  return __builtin_amdgcn_wmma_f32_16x16x32_bf16(false, a, false, b, (short)0, c,
                                                 false, false);
}

// A fragment (16x32, M x K), row-major source [*, lda] bf16, tile origin (m0,k0).
// ISA layout: lanes 0-15 hold M=lane, K=[0..7]+[16..23]; lanes 16-31 K=[8..15]+[24..31].
static __device__ __forceinline__ v16bf load_frag_a(const __bf16* A, int lda,
                                                    int m0, int k0) {
  int lane = threadIdx.x & 31;
  int row  = m0 + (lane & 15);
  int koff = k0 + ((lane >> 4) << 3);
  const __bf16* p = A + (size_t)row * lda + koff;
  v8bf lo = *(const v8bf*)p;          // K = koff .. koff+7
  v8bf hi = *(const v8bf*)(p + 16);   // K = koff+16 .. koff+23
  return make16(lo, hi);
}

// B fragment (32x16, K x N) of B = W^T, W row-major [N, ldb] bf16, tile (k0,n0).
// Lanes 0-15: col N=lane, K=[0..7]+[16..23]; lanes 16-31: K=[8..15]+[24..31].
static __device__ __forceinline__ v16bf load_frag_b(const __bf16* W, int ldb,
                                                    int n0, int k0) {
  int lane = threadIdx.x & 31;
  int coln = n0 + (lane & 15);
  int koff = k0 + ((lane >> 4) << 3);
  const __bf16* p = W + (size_t)coln * ldb + koff;
  v8bf lo = *(const v8bf*)p;
  v8bf hi = *(const v8bf*)(p + 16);
  return make16(lo, hi);
}

// ---------- fp32 -> bf16 conversion ----------
__global__ void cvt_bf16_kernel(const float* __restrict__ in,
                                __bf16* __restrict__ out, int n) {
  int i = blockIdx.x * blockDim.x + threadIdx.x;
  if (i < n) out[i] = f2bf(in[i]);
}

// ---------- QKV projection GEMM: [4096,1024] x [3072,1024]^T ----------
// 4 waves/block; wave w computes a 32(M) x 64(N) tile: 2 A frags x 4 B frags
// -> 8 independent WMMAs per k-step; all loads hoisted ahead of the WMMA chain
// so the backend can issue partial s_wait_loadcnt and overlap loads with math.
__global__ void __launch_bounds__(128) qkv_gemm_kernel(
    const __bf16* __restrict__ A, const __bf16* __restrict__ W,
    const float* __restrict__ bias,
    __bf16* __restrict__ q, __bf16* __restrict__ k, __bf16* __restrict__ vT) {
  const int wave = threadIdx.x >> 5;
  const int lane = threadIdx.x & 31;
  const int col  = lane & 15;
  const int half = (lane >> 4) & 1;
  const int m0 = blockIdx.x * 128 + wave * 32;
  const int n0 = blockIdx.y * 64;
  v8f acc[2][4] = {};
  for (int k0 = 0; k0 < DD; k0 += 32) {
    v16bf a0 = load_frag_a(A, DD, m0,      k0);
    v16bf a1 = load_frag_a(A, DD, m0 + 16, k0);
    v16bf b0 = load_frag_b(W, DD, n0,      k0);
    v16bf b1 = load_frag_b(W, DD, n0 + 16, k0);
    v16bf b2 = load_frag_b(W, DD, n0 + 32, k0);
    v16bf b3 = load_frag_b(W, DD, n0 + 48, k0);
    if (k0 + 32 < DD)
      __builtin_prefetch((const void*)(A + (size_t)(m0 + col) * DD + k0 + 32), 0, 3);
    acc[0][0] = wmma_bf16(a0, b0, acc[0][0]);
    acc[0][1] = wmma_bf16(a0, b1, acc[0][1]);
    acc[0][2] = wmma_bf16(a0, b2, acc[0][2]);
    acc[0][3] = wmma_bf16(a0, b3, acc[0][3]);
    acc[1][0] = wmma_bf16(a1, b0, acc[1][0]);
    acc[1][1] = wmma_bf16(a1, b1, acc[1][1]);
    acc[1][2] = wmma_bf16(a1, b2, acc[1][2]);
    acc[1][3] = wmma_bf16(a1, b3, acc[1][3]);
  }
#pragma unroll
  for (int mi = 0; mi < 2; mi++) {
#pragma unroll
    for (int nt = 0; nt < 4; nt++) {
#pragma unroll
      for (int r = 0; r < 8; r++) {
        int m = m0 + mi * 16 + r + 8 * half;       // flat row in [0, B*S)
        int n = n0 + nt * 16 + col;                // col in [0, 3*D)
        float v = acc[mi][nt][r] + bias[n];
        int b   = m >> 11;                         // / S
        int s   = m & (SS - 1);
        int wh  = n >> 10;                         // 0=q 1=k 2=v
        int rem = n & (DD - 1);
        int h   = rem >> 6;
        int dm  = rem & 63;
        size_t hb = (size_t)(b * HH + h);
        if (wh == 0)      q [(hb * SS + s) * HDIM + dm] = f2bf(v * 0.125f); // fold 1/sqrt(64)
        else if (wh == 1) k [(hb * SS + s) * HDIM + dm] = f2bf(v);
        else              vT[(hb * HDIM + dm) * SS + s] = f2bf(v);          // V transposed
      }
    }
  }
}

// ---------- Flash attention (causal), one (b,h,64-row q tile) per block ----------
__global__ void __launch_bounds__(128) flash_attn_kernel(
    const __bf16* __restrict__ qb, const __bf16* __restrict__ kb,
    const __bf16* __restrict__ vb, __bf16* __restrict__ ctx) {
  __shared__ __bf16 plds[4 * 16 * 32];           // 1 KB per wave P staging
  const int wave = threadIdx.x >> 5;
  const int lane = threadIdx.x & 31;
  const int col  = lane & 15;
  const int half = (lane >> 4) & 1;
  int blk   = blockIdx.x;
  int qtile = blk & 31;                          // S/64 = 32 tiles
  int bh    = blk >> 5;
  int b = bh >> 4, h = bh & 15;
  const int qt = qtile * 64 + wave * 16;         // this wave's 16 query rows
  size_t hb = (size_t)(b * HH + h);
  const __bf16* Q  = qb + hb * SS * HDIM;
  const __bf16* K  = kb + hb * SS * HDIM;
  const __bf16* Vt = vb + hb * HDIM * SS;        // [64][S]
  __bf16* wl = plds + wave * 512;

  v16bf qa0 = load_frag_a(Q, HDIM, qt, 0);
  v16bf qa1 = load_frag_a(Q, HDIM, qt, 32);
  float m_i[8], l_i[8];
#pragma unroll
  for (int r = 0; r < 8; r++) { m_i[r] = -3.0e38f; l_i[r] = 0.0f; }
  v8f o[4] = {};

  for (int j0 = 0; j0 < qt + 16; j0 += 32) {     // causal: key blocks up to qt+15
    // hoist all 4 K-fragment loads before the WMMA chain
    v16bf k00 = load_frag_b(K, HDIM, j0,      0);
    v16bf k01 = load_frag_b(K, HDIM, j0,      32);
    v16bf k10 = load_frag_b(K, HDIM, j0 + 16, 0);
    v16bf k11 = load_frag_b(K, HDIM, j0 + 16, 32);
    v8f s0 = {}, s1 = {};
    s0 = wmma_bf16(qa0, k00, s0);
    s0 = wmma_bf16(qa1, k01, s0);
    s1 = wmma_bf16(qa0, k10, s1);
    s1 = wmma_bf16(qa1, k11, s1);

    float corr[8], p0a[8], p1a[8];
#pragma unroll
    for (int r = 0; r < 8; r++) {
      int mrow = qt + r + 8 * half;
      float v0 = s0[r]; if (j0 + col > mrow)      v0 = -1.0e30f;  // causal mask
      float v1 = s1[r]; if (j0 + 16 + col > mrow) v1 = -1.0e30f;
      float tmax = fmaxf(v0, v1);                 // row-max across 16 lanes
      tmax = fmaxf(tmax, __shfl_xor(tmax, 1, 32));
      tmax = fmaxf(tmax, __shfl_xor(tmax, 2, 32));
      tmax = fmaxf(tmax, __shfl_xor(tmax, 4, 32));
      tmax = fmaxf(tmax, __shfl_xor(tmax, 8, 32));
      float nm = fmaxf(m_i[r], tmax);
      float c  = __expf(m_i[r] - nm);
      float p0 = __expf(v0 - nm);
      float p1 = __expf(v1 - nm);
      float rs = p0 + p1;
      rs += __shfl_xor(rs, 1, 32);
      rs += __shfl_xor(rs, 2, 32);
      rs += __shfl_xor(rs, 4, 32);
      rs += __shfl_xor(rs, 8, 32);
      l_i[r] = l_i[r] * c + rs;
      m_i[r] = nm;
      corr[r] = c; p0a[r] = p0; p1a[r] = p1;
    }
#pragma unroll
    for (int nt = 0; nt < 4; nt++)
#pragma unroll
      for (int r = 0; r < 8; r++) o[nt][r] *= corr[r];

    // C-layout -> A-layout redistribution of P through per-wave LDS
#pragma unroll
    for (int r = 0; r < 8; r++) {
      int rl = r + 8 * half;
      wl[rl * 32 + col]      = f2bf(p0a[r]);
      wl[rl * 32 + 16 + col] = f2bf(p1a[r]);
    }
    asm volatile("s_wait_dscnt 0x0" ::: "memory");   // CDNA5 split DS counter wait
    const __bf16* pp = wl + col * 32 + 8 * half;
    v8bf lo = *(const v8bf*)pp;
    v8bf hi = *(const v8bf*)(pp + 16);
    v16bf pf = make16(lo, hi);
    // hoist all 4 V-fragment loads before the WMMA chain
    v16bf vf0 = load_frag_b(Vt, SS, 0,  j0);
    v16bf vf1 = load_frag_b(Vt, SS, 16, j0);
    v16bf vf2 = load_frag_b(Vt, SS, 32, j0);
    v16bf vf3 = load_frag_b(Vt, SS, 48, j0);
    o[0] = wmma_bf16(pf, vf0, o[0]);
    o[1] = wmma_bf16(pf, vf1, o[1]);
    o[2] = wmma_bf16(pf, vf2, o[2]);
    o[3] = wmma_bf16(pf, vf3, o[3]);
  }

#pragma unroll
  for (int r = 0; r < 8; r++) {
    float inv = 1.0f / l_i[r];
    int srow = qt + r + 8 * half;
    size_t mrow = (size_t)b * SS + srow;
#pragma unroll
    for (int nt = 0; nt < 4; nt++) {
      int cd = h * HDIM + nt * 16 + col;
      ctx[mrow * DD + cd] = f2bf(o[nt][r] * inv);
    }
  }
}

// ---------- Output projection GEMM: [4096,1024] x [1024,1024]^T -> fp32 ----------
__global__ void __launch_bounds__(128) out_gemm_kernel(
    const __bf16* __restrict__ A, const __bf16* __restrict__ W,
    const float* __restrict__ bias, float* __restrict__ C) {
  const int wave = threadIdx.x >> 5;
  const int lane = threadIdx.x & 31;
  const int col  = lane & 15;
  const int half = (lane >> 4) & 1;
  const int m0 = blockIdx.x * 128 + wave * 32;
  const int n0 = blockIdx.y * 64;
  v8f acc[2][4] = {};
  for (int k0 = 0; k0 < DD; k0 += 32) {
    v16bf a0 = load_frag_a(A, DD, m0,      k0);
    v16bf a1 = load_frag_a(A, DD, m0 + 16, k0);
    v16bf b0 = load_frag_b(W, DD, n0,      k0);
    v16bf b1 = load_frag_b(W, DD, n0 + 16, k0);
    v16bf b2 = load_frag_b(W, DD, n0 + 32, k0);
    v16bf b3 = load_frag_b(W, DD, n0 + 48, k0);
    if (k0 + 32 < DD)
      __builtin_prefetch((const void*)(A + (size_t)(m0 + col) * DD + k0 + 32), 0, 3);
    acc[0][0] = wmma_bf16(a0, b0, acc[0][0]);
    acc[0][1] = wmma_bf16(a0, b1, acc[0][1]);
    acc[0][2] = wmma_bf16(a0, b2, acc[0][2]);
    acc[0][3] = wmma_bf16(a0, b3, acc[0][3]);
    acc[1][0] = wmma_bf16(a1, b0, acc[1][0]);
    acc[1][1] = wmma_bf16(a1, b1, acc[1][1]);
    acc[1][2] = wmma_bf16(a1, b2, acc[1][2]);
    acc[1][3] = wmma_bf16(a1, b3, acc[1][3]);
  }
#pragma unroll
  for (int mi = 0; mi < 2; mi++) {
#pragma unroll
    for (int nt = 0; nt < 4; nt++) {
#pragma unroll
      for (int r = 0; r < 8; r++) {
        int m = m0 + mi * 16 + r + 8 * half;
        int n = n0 + nt * 16 + col;
        C[(size_t)m * DD + n] = acc[mi][nt][r] + bias[n];
      }
    }
  }
}

// ---------- sel logits: one wave per (b,s) row ----------
__global__ void __launch_bounds__(128) sel_logits_kernel(
    const float* __restrict__ x, const float* __restrict__ sw,
    const float* __restrict__ sb, float* __restrict__ sel) {
  int wave = threadIdx.x >> 5;
  int lane = threadIdx.x & 31;
  int row = blockIdx.x * 4 + wave;
  const float* xr = x + (size_t)row * DD;
  float acc = 0.f;
  for (int i = lane; i < DD; i += 32) acc += xr[i] * sw[i];
#pragma unroll
  for (int m = 1; m < 32; m <<= 1) acc += __shfl_xor(acc, m, 32);
  if (lane == 0) sel[row] = acc + sb[0];
}

// ---------- softmax over sequence (per batch), * sqrt(D)/8 = 4 ----------
__global__ void __launch_bounds__(256) sel_softmax_kernel(float* __restrict__ sel) {
  __shared__ float red[256];
  float* p = sel + (size_t)blockIdx.x * SS;
  int t = threadIdx.x;
  float mx = -3.0e38f;
  for (int i = t; i < SS; i += 256) mx = fmaxf(mx, p[i]);
  red[t] = mx; __syncthreads();
  for (int s = 128; s > 0; s >>= 1) { if (t < s) red[t] = fmaxf(red[t], red[t + s]); __syncthreads(); }
  mx = red[0]; __syncthreads();
  float sum = 0.f;
  for (int i = t; i < SS; i += 256) sum += __expf(p[i] - mx);
  red[t] = sum; __syncthreads();
  for (int s = 128; s > 0; s >>= 1) { if (t < s) red[t] += red[t + s]; __syncthreads(); }
  sum = red[0]; __syncthreads();
  float scale = 4.0f / sum;
  for (int i = t; i < SS; i += 256) p[i] = __expf(p[i] - mx) * scale;
}

__global__ void zero_flags_kernel(int* __restrict__ flags, int n) {
  int i = blockIdx.x * blockDim.x + threadIdx.x;
  if (i < n) flags[i] = 0;
}

// ---------- top-k (k=46) per batch; smallest-index tie-breaking like lax.top_k ----------
__global__ void __launch_bounds__(256) topk_kernel(const float* __restrict__ sel,
                                                   int* __restrict__ flags) {
  __shared__ float vals[SS];
  __shared__ float rv[256];
  __shared__ int   ri[256];
  int b = blockIdx.x, t = threadIdx.x;
  const float* p = sel + (size_t)b * SS;
  for (int i = t; i < SS; i += 256) vals[i] = p[i];
  __syncthreads();
  for (int r = 0; r < KTOP; r++) {
    float bv = -3.0e38f; int bi = SS;
    for (int i = t; i < SS; i += 256) {
      float v = vals[i];
      if (v > bv) { bv = v; bi = i; }           // strided ascending -> first max kept
    }
    rv[t] = bv; ri[t] = bi; __syncthreads();
    for (int s = 128; s > 0; s >>= 1) {
      if (t < s) {
        if (rv[t + s] > rv[t] || (rv[t + s] == rv[t] && ri[t + s] < ri[t])) {
          rv[t] = rv[t + s]; ri[t] = ri[t + s];
        }
      }
      __syncthreads();
    }
    if (t == 0) { flags[b * SS + ri[0]] = 1; vals[ri[0]] = -3.0e38f; }
    __syncthreads();
  }
}

// ---------- y = x + flag * (attn_out * sel) ----------
__global__ void final_kernel(const float* __restrict__ x, const float* __restrict__ attn,
                             const float* __restrict__ sel, const int* __restrict__ flags,
                             float* __restrict__ y, size_t total) {
  size_t i = (size_t)blockIdx.x * blockDim.x + threadIdx.x;
  if (i >= total) return;
  size_t row = i / DD;
  float v = x[i];
  if (flags[row]) v += attn[i] * sel[row];
  y[i] = v;
}

extern "C" void kernel_launch(void* const* d_in, const int* in_sizes, int n_in,
                              void* d_out, int out_size, void* d_ws, size_t ws_size,
                              hipStream_t stream) {
  (void)in_sizes; (void)n_in; (void)out_size; (void)ws_size;
  const float* x      = (const float*)d_in[0];
  const float* Wqkv_w = (const float*)d_in[1];
  const float* Wqkv_b = (const float*)d_in[2];
  const float* sel_w  = (const float*)d_in[3];
  const float* sel_b  = (const float*)d_in[4];
  const float* out_w  = (const float*)d_in[5];
  const float* out_b  = (const float*)d_in[6];
  float* y = (float*)d_out;

  char* ws = (char*)d_ws;
  size_t off = 0;
  auto alloc = [&](size_t bytes) -> void* {
    void* p = (void*)(ws + off);
    off += (bytes + 255) & ~(size_t)255;
    return p;
  };
  __bf16* xbf   = (__bf16*)alloc((size_t)BB * SS * DD * 2);
  __bf16* wqkv  = (__bf16*)alloc((size_t)3 * DD * DD * 2);
  __bf16* wout  = (__bf16*)alloc((size_t)DD * DD * 2);
  __bf16* qb    = (__bf16*)alloc((size_t)BB * HH * SS * HDIM * 2);
  __bf16* kb    = (__bf16*)alloc((size_t)BB * HH * SS * HDIM * 2);
  __bf16* vb    = (__bf16*)alloc((size_t)BB * HH * SS * HDIM * 2);
  __bf16* ctx   = (__bf16*)alloc((size_t)BB * SS * DD * 2);
  float*  attn  = (float*)alloc((size_t)BB * SS * DD * 4);
  float*  sel   = (float*)alloc((size_t)BB * SS * 4);
  int*    flags = (int*)alloc((size_t)BB * SS * 4);

  const int nX = BB * SS * DD;
  cvt_bf16_kernel<<<(nX + 255) / 256, 256, 0, stream>>>(x, xbf, nX);
  cvt_bf16_kernel<<<(3 * DD * DD + 255) / 256, 256, 0, stream>>>(Wqkv_w, wqkv, 3 * DD * DD);
  cvt_bf16_kernel<<<(DD * DD + 255) / 256, 256, 0, stream>>>(out_w, wout, DD * DD);

  dim3 g1((BB * SS) / 128, (3 * DD) / 64);       // 32 x 48
  qkv_gemm_kernel<<<g1, 128, 0, stream>>>(xbf, wqkv, Wqkv_b, qb, kb, vb);

  flash_attn_kernel<<<BB * HH * (SS / 64), 128, 0, stream>>>(qb, kb, vb, ctx);

  dim3 g2((BB * SS) / 128, DD / 64);             // 32 x 16
  out_gemm_kernel<<<g2, 128, 0, stream>>>(ctx, wout, out_b, attn);

  sel_logits_kernel<<<(BB * SS) / 4, 128, 0, stream>>>(x, sel_w, sel_b, sel);
  sel_softmax_kernel<<<BB, 256, 0, stream>>>(sel);
  zero_flags_kernel<<<(BB * SS + 255) / 256, 256, 0, stream>>>(flags, BB * SS);
  topk_kernel<<<BB, 256, 0, stream>>>(sel, flags);
  final_kernel<<<(nX + 255) / 256, 256, 0, stream>>>(x, attn, sel, flags, y, (size_t)nX);
}
